// MethodCFGEncoder_47184510714131
// MI455X (gfx1250) — matrix operand order repaired and checked by reading, not compiled, not run on hardware
//
#include <hip/hip_runtime.h>
#include <hip/hip_bf16.h>

#define D 256
#define E 512
#define N_NODES 20000
#define N_EXPR 8000
#define T_EXPR 32
#define N_PATHS 2000
#define PATH_LEN 32
#define N_SYM 6000
#define N_OCC 60000

typedef __bf16 bf16_t;
typedef __attribute__((ext_vector_type(16))) __bf16 v16bf;
typedef __attribute__((ext_vector_type(8)))  float  v8f;

// ---------------------------------------------------------------------------
// WMMA fragment layouts (CDNA5 ISA 7.12.2, wave32):
// A (16x32 bf16): lane = (khalf<<4)|m ; 8 VGPRs x 2 elems ; per lane elems e:
//   e in [0,8)  -> k = 8*khalf + e
//   e in [8,16) -> k = 16 + 8*khalf + (e-8)
// B (32x16 bf16): lane = (kgrp<<4)|n ; per lane elem e: same k order as A
// C/D (16x16 f32): reg r, lane -> m = r + 8*(lane>>4), n = lane&15
// We pre-swizzle A into LDS and B into workspace so each lane's 16 bf16
// fragment is contiguous (32B) -> b128 loads.
// ---------------------------------------------------------------------------

__device__ __forceinline__ int afrag_pos(int m, int k) {
  int ktile = k >> 5, kk = k & 31;
  int g = (kk >> 3) & 1;                       // 0..7 ->0, 8..15 ->1, 16..23->0, 24..31->1
  int e = (kk & 7) + ((kk & 16) ? 8 : 0);
  int lane = (g << 4) | m;
  return ((ktile << 5) + lane) * 16 + e;
}

__device__ __forceinline__ int bfrag_pos(int k, int n, int K) {
  int ktile = k >> 5, kk = k & 31;
  int g = (kk >> 3) & 1, e = (kk & 7) + ((kk & 16) ? 8 : 0);
  int lane = (g << 4) | (n & 15);
  int ntile = n >> 4;
  return ((ntile * (K >> 5) + ktile) * 32 + lane) * 16 + e;
}

__device__ __forceinline__ v8f wmma_bf16(v16bf a, v16bf b, v8f c) {
  return __builtin_amdgcn_wmma_f32_16x16x32_bf16(
      /*neg_a=*/false, a, /*neg_b=*/false, b,
      /*c_mod=*/(short)0, c, /*reuse_a=*/false, /*reuse_b=*/false);
}

__device__ __forceinline__ unsigned int f2ord(float f) {
  unsigned int u = __float_as_uint(f);
  return (u & 0x80000000u) ? ~u : (u | 0x80000000u);
}
__device__ __forceinline__ float ord2f(unsigned int u) {
  u = (u & 0x80000000u) ? (u & 0x7FFFFFFFu) : ~u;
  return __uint_as_float(u);
}

// ---------------------------------------------------------------------------
// Utility kernels
// ---------------------------------------------------------------------------
__global__ void k_copy_f4(float4* __restrict__ dst, const float4* __restrict__ src, long n4) {
  long i = blockIdx.x * (long)blockDim.x + threadIdx.x;
  long stride = (long)gridDim.x * blockDim.x;
  for (; i < n4; i += stride) dst[i] = src[i];
}

__global__ void k_zero_f32(float* __restrict__ p, long n) {
  long i = blockIdx.x * (long)blockDim.x + threadIdx.x;
  long stride = (long)gridDim.x * blockDim.x;
  for (; i < n; i += stride) p[i] = 0.0f;
}

__global__ void k_init_seg(unsigned int* __restrict__ smax, float* __restrict__ denom, int n) {
  int i = blockIdx.x * blockDim.x + threadIdx.x;
  if (i < n) { smax[i] = 0x007FFFFFu /* = f2ord(-inf) */; denom[i] = 0.0f; }
}

// Pack a row-major KxN f32 weight matrix into bf16 B-fragment order.
__global__ void k_pack_w(const float* __restrict__ src, bf16_t* __restrict__ dst, int K, int N) {
  int total = K * N;
  int i = blockIdx.x * blockDim.x + threadIdx.x;
  int stride = gridDim.x * blockDim.x;
  for (; i < total; i += stride) {
    int k = i / N, n = i - k * N;
    dst[bfrag_pos(k, n, K)] = (bf16_t)src[i];
  }
}

// ---------------------------------------------------------------------------
// Stage 1: new_enc = relu([gather(prev_nodes), comb] @ W_proj + b), scattered.
// Block: 256 threads (8 waves), 16 rows x 256 cols per block. K = 768.
// ---------------------------------------------------------------------------
__global__ __launch_bounds__(256) void k_update_nodes(
    const float* __restrict__ prev_nodes, const float* __restrict__ comb,
    const int* __restrict__ gidx, const bf16_t* __restrict__ WprojPk,
    const float* __restrict__ b_proj, float* __restrict__ nodes) {
  __shared__ __attribute__((aligned(32))) bf16_t afrag[16 * 768];
  const int tid = threadIdx.x, lane = tid & 31, wave = tid >> 5;
  const int mbase = blockIdx.x * 16;

  for (int e = tid; e < 16 * 768; e += 256) {
    int m = e / 768, k = e - m * 768;
    int row = mbase + m;
    float v = (k < 256) ? prev_nodes[(long)gidx[row] * 256 + k]
                        : comb[(long)row * 512 + (k - 256)];
    afrag[afrag_pos(m, k)] = (bf16_t)v;
  }
  __syncthreads();

  for (int nt = wave * 2; nt < wave * 2 + 2; ++nt) {
    v8f acc = {};
    for (int kt = 0; kt < 24; ++kt) {
      v16bf a = *(const v16bf*)&afrag[(kt * 32 + lane) * 16];
      v16bf b = *(const v16bf*)&WprojPk[((nt * 24 + kt) * 32 + lane) * 16];
      acc = wmma_bf16(a, b, acc);
    }
    int n = nt * 16 + (lane & 15);
    int mo = (lane >> 4) * 8;
#pragma unroll
    for (int r = 0; r < 8; ++r) {
      int row = mbase + mo + r;
      float v = acc[r] + b_proj[n];
      nodes[(long)gidx[row] * 256 + n] = v > 0.0f ? v : 0.0f;
    }
  }
}

// ---------------------------------------------------------------------------
// Stage 2: GRU over 64 interleaved (node, edge) tokens, 16 paths per block.
// 512 threads = 16 waves; wave w owns hidden columns [16w, 16w+16).
// Per step, per wave: 6 accumulators (r/z/n for x@W_ih and h@W_hh),
// 48 v_wmma_f32_16x16x32_bf16; gating done in-register on the C/D layout.
// Only even-step (node-position) hidden states are written to `scat`.
// ---------------------------------------------------------------------------
__global__ __launch_bounds__(512) void k_gru(
    const float* __restrict__ pno, const float* __restrict__ peo,
    const int* __restrict__ pidx, const float* __restrict__ nodes,
    const bf16_t* __restrict__ WihPk, const bf16_t* __restrict__ WhhPk,
    const float* __restrict__ b_ih, const float* __restrict__ b_hh,
    float* __restrict__ scat) {
  __shared__ __attribute__((aligned(32))) bf16_t xfrag[16 * 256];
  __shared__ __attribute__((aligned(32))) bf16_t hfrag[16 * 256];
  __shared__ float hbuf[16 * 256];

  const int tid = threadIdx.x, lane = tid & 31, wave = tid >> 5;
  const int pbase = blockIdx.x * 16;
  const int dcol = wave * 16 + (lane & 15);
  const int mo = (lane >> 4) * 8;
  const int ntr = wave, ntz = wave + 16, ntn = wave + 32;

  for (int e = tid; e < 16 * 256; e += 512) { hbuf[e] = 0.0f; hfrag[e] = (bf16_t)0.0f; }
  __syncthreads();

  const float bir = b_ih[dcol],        bhr = b_hh[dcol];
  const float biz = b_ih[256 + dcol],  bhz = b_hh[256 + dcol];
  const float bin_ = b_ih[512 + dcol], bhn = b_hh[512 + dcol];

  for (int t = 0; t < 2 * PATH_LEN; ++t) {
    const int j = t >> 1;
    // Phase 1: stage x_t (gathered for node steps) as bf16 A-fragments.
    for (int e = tid; e < 16 * 256; e += 512) {
      int m = e >> 8, k = e & 255;
      long occ = (long)(pbase + m) * PATH_LEN + j;
      float v;
      if ((t & 1) == 0) {
        int nidx = pidx[occ];
        v = pno[occ * 256 + k] + nodes[(long)nidx * 256 + k];
      } else {
        v = peo[occ * 256 + k];
      }
      xfrag[afrag_pos(m, k)] = (bf16_t)v;
    }
    __syncthreads();

    // Phase 2: 6 gate GEMM tiles, K = 256 (8 WMMA steps each).
    v8f axr = {}, axz = {}, axn = {}, ahr = {}, ahz = {}, ahn = {};
    for (int kt = 0; kt < 8; ++kt) {
      v16bf ax = *(const v16bf*)&xfrag[(kt * 32 + lane) * 16];
      v16bf ah = *(const v16bf*)&hfrag[(kt * 32 + lane) * 16];
      v16bf b0 = *(const v16bf*)&WihPk[((ntr * 8 + kt) * 32 + lane) * 16];
      v16bf b1 = *(const v16bf*)&WihPk[((ntz * 8 + kt) * 32 + lane) * 16];
      v16bf b2 = *(const v16bf*)&WihPk[((ntn * 8 + kt) * 32 + lane) * 16];
      axr = wmma_bf16(ax, b0, axr);
      axz = wmma_bf16(ax, b1, axz);
      axn = wmma_bf16(ax, b2, axn);
      b0 = *(const v16bf*)&WhhPk[((ntr * 8 + kt) * 32 + lane) * 16];
      b1 = *(const v16bf*)&WhhPk[((ntz * 8 + kt) * 32 + lane) * 16];
      b2 = *(const v16bf*)&WhhPk[((ntn * 8 + kt) * 32 + lane) * 16];
      ahr = wmma_bf16(ah, b0, ahr);
      ahz = wmma_bf16(ah, b1, ahz);
      ahn = wmma_bf16(ah, b2, ahn);
    }
    __syncthreads();  // all waves done reading x/h fragments

    // Phase 3: elementwise gating directly on accumulator registers.
#pragma unroll
    for (int r = 0; r < 8; ++r) {
      int m = mo + r;
      float rg = 1.0f / (1.0f + __expf(-((axr[r] + bir) + (ahr[r] + bhr))));
      float zg = 1.0f / (1.0f + __expf(-((axz[r] + biz) + (ahz[r] + bhz))));
      float ng = tanhf((axn[r] + bin_) + rg * (ahn[r] + bhn));
      float hold = hbuf[m * 256 + dcol];
      float hnew = (1.0f - zg) * ng + zg * hold;
      hbuf[m * 256 + dcol] = hnew;
      hfrag[afrag_pos(m, dcol)] = (bf16_t)hnew;
      if ((t & 1) == 0) {
        long occ = (long)(pbase + m) * PATH_LEN + j;
        scat[occ * 256 + dcol] = hnew;
      }
    }
    __syncthreads();  // h updates visible before next step
  }
}

// ---------------------------------------------------------------------------
// Stage 3a: q = nodes @ W_q  (20000x256 @ 256x256)
// ---------------------------------------------------------------------------
__global__ __launch_bounds__(256) void k_gemm_q(
    const float* __restrict__ nodes, const bf16_t* __restrict__ WqPk,
    float* __restrict__ q) {
  __shared__ __attribute__((aligned(32))) bf16_t afrag[16 * 256];
  const int tid = threadIdx.x, lane = tid & 31, wave = tid >> 5;
  const int mbase = blockIdx.x * 16;

  for (int e = tid; e < 16 * 256; e += 256) {
    int m = e >> 8, k = e & 255;
    afrag[afrag_pos(m, k)] = (bf16_t)nodes[(long)(mbase + m) * 256 + k];
  }
  __syncthreads();

  for (int nt = wave * 2; nt < wave * 2 + 2; ++nt) {
    v8f acc = {};
    for (int kt = 0; kt < 8; ++kt) {
      v16bf a = *(const v16bf*)&afrag[(kt * 32 + lane) * 16];
      v16bf b = *(const v16bf*)&WqPk[((nt * 8 + kt) * 32 + lane) * 16];
      acc = wmma_bf16(a, b, acc);
    }
    int n = nt * 16 + (lane & 15);
    int mo = (lane >> 4) * 8;
#pragma unroll
    for (int r = 0; r < 8; ++r) q[(long)(mbase + mo + r) * 256 + n] = acc[r];
  }
}

// ---------------------------------------------------------------------------
// Stage 3b: per-occurrence scores + segment max (order-preserving uint max).
// One wave per row, shuffle reduction.
// ---------------------------------------------------------------------------
__global__ __launch_bounds__(256) void k_scores(
    const float* __restrict__ scat, const float* __restrict__ q,
    const int* __restrict__ pidx, const unsigned char* __restrict__ mask,
    float* __restrict__ scores, unsigned int* __restrict__ smax_bits) {
  int row = blockIdx.x * 8 + (threadIdx.x >> 5);
  int lane = threadIdx.x & 31;
  if (row >= N_PATHS * PATH_LEN) return;
  int seg = pidx[row];
  bool ok = mask[row] != 0;
  const float4* s4 = (const float4*)(scat + (long)row * 256);
  const float4* q4 = (const float4*)(q + (long)seg * 256);
  float s = 0.0f;
  for (int i = lane; i < 64; i += 32) {
    float4 a = s4[i], b = q4[i];
    s += a.x * b.x + a.y * b.y + a.z * b.z + a.w * b.w;
  }
#pragma unroll
  for (int off = 16; off > 0; off >>= 1) s += __shfl_down(s, off, 32);
  if (lane == 0) {
    s *= 0.0625f;  // 1/sqrt(256)
    scores[row] = s;
    if (ok) atomicMax(&smax_bits[seg], f2ord(s));
  }
}

__global__ void k_soft1(const int* __restrict__ pidx, const unsigned char* __restrict__ mask,
                        const unsigned int* __restrict__ smax_bits,
                        float* __restrict__ scores, float* __restrict__ denom) {
  int row = blockIdx.x * blockDim.x + threadIdx.x;
  if (row >= N_PATHS * PATH_LEN) return;
  int seg = pidx[row];
  bool ok = mask[row] != 0;
  float smax = ord2f(smax_bits[seg]);
  if (!(smax > -1e37f && smax < 1e37f)) smax = 0.0f;
  float ex = ok ? __expf(scores[row] - smax) : 0.0f;
  scores[row] = ex;
  if (ok) atomicAdd(&denom[seg], ex);
}

__global__ __launch_bounds__(256) void k_soft2(
    const float* __restrict__ scat, const int* __restrict__ pidx,
    const unsigned char* __restrict__ mask, const float* __restrict__ scores,
    const float* __restrict__ denom, float* __restrict__ nodes_out) {
  int row = blockIdx.x;
  if (!mask[row]) return;
  int seg = pidx[row];
  if (seg >= N_NODES) return;
  float alpha = scores[row] / fmaxf(denom[seg], 1e-9f);
  int d2 = threadIdx.x;
  atomicAdd(&nodes_out[(long)seg * 256 + d2], alpha * scat[(long)row * 256 + d2]);
}

// ---------------------------------------------------------------------------
// Stage 4: expression symbol scatter-add.
// ---------------------------------------------------------------------------
__global__ __launch_bounds__(256) void k_sym_scatter(
    const int* __restrict__ sae, const int* __restrict__ sat,
    const int* __restrict__ sas, const float* __restrict__ syms,
    float* __restrict__ expr_out) {
  int o = blockIdx.x;
  long fi = (long)sae[o] * T_EXPR + sat[o];
  atomicAdd(&expr_out[fi * 256 + threadIdx.x], syms[(long)sas[o] * 256 + threadIdx.x]);
}

// ---------------------------------------------------------------------------
// Host launcher
// ---------------------------------------------------------------------------
extern "C" void kernel_launch(void* const* d_in, const int* in_sizes, int n_in,
                              void* d_out, int out_size, void* d_ws, size_t ws_size,
                              hipStream_t stream) {
  const float* expr_enc   = (const float*)d_in[0];   // 8000x32x256
  const float* syms       = (const float*)d_in[1];   // 6000x256
  const float* prev_nodes = (const float*)d_in[2];   // 20000x256
  const float* comb       = (const float*)d_in[3];   // 8000x512
  const float* pno        = (const float*)d_in[4];   // 2000x32x256
  const float* peo        = (const float*)d_in[5];   // 2000x32x256
  const int*   gidx       = (const int*)d_in[6];     // 8000
  const int*   pidx       = (const int*)d_in[7];     // 2000x32
  const unsigned char* mask = (const unsigned char*)d_in[8]; // 2000x32 bool
  const int*   sae        = (const int*)d_in[9];     // 60000
  const int*   sat        = (const int*)d_in[10];
  const int*   sas        = (const int*)d_in[11];
  const float* W_proj     = (const float*)d_in[12];  // 768x256
  const float* b_proj     = (const float*)d_in[13];
  const float* W_ih       = (const float*)d_in[14];  // 256x768
  const float* W_hh       = (const float*)d_in[15];  // 256x768
  const float* b_ih       = (const float*)d_in[16];
  const float* b_hh       = (const float*)d_in[17];
  const float* W_q        = (const float*)d_in[18];  // 256x256

  const long EXPR_ELEMS  = (long)N_EXPR * T_EXPR * D;   // 65,536,000
  const long NODES_ELEMS = (long)N_NODES * D;           //  5,120,000
  float* expr_out  = (float*)d_out;
  float* nodes_out = (float*)d_out + EXPR_ELEMS;

  // Workspace layout (bytes, 256-aligned)
  char* ws = (char*)d_ws;
  size_t off = 0;
  auto take = [&](size_t bytes) { char* p = ws + off; off = (off + bytes + 255) & ~(size_t)255; return p; };
  float*        nodes_ws  = (float*)take(NODES_ELEMS * 4);
  float*        q_ws      = (float*)take(NODES_ELEMS * 4);
  float*        scat      = (float*)take((long)N_PATHS * PATH_LEN * D * 4);
  float*        scores    = (float*)take((long)N_PATHS * PATH_LEN * 4);
  unsigned int* smax_bits = (unsigned int*)take((N_NODES + 1) * 4);
  float*        denom     = (float*)take((N_NODES + 1) * 4);
  bf16_t*       WprojPk   = (bf16_t*)take((long)768 * 256 * 2);
  bf16_t*       WihPk     = (bf16_t*)take((long)256 * 768 * 2);
  bf16_t*       WhhPk     = (bf16_t*)take((long)256 * 768 * 2);
  bf16_t*       WqPk      = (bf16_t*)take((long)256 * 256 * 2);
  (void)ws_size; (void)in_sizes; (void)n_in; (void)out_size;

  // Init / copies
  k_copy_f4<<<8192, 256, 0, stream>>>((float4*)expr_out, (const float4*)expr_enc, EXPR_ELEMS / 4);
  k_copy_f4<<<2048, 256, 0, stream>>>((float4*)nodes_ws, (const float4*)prev_nodes, NODES_ELEMS / 4);
  k_zero_f32<<<2048, 256, 0, stream>>>(nodes_out, NODES_ELEMS);
  k_init_seg<<<(N_NODES + 1 + 255) / 256, 256, 0, stream>>>(smax_bits, denom, N_NODES + 1);

  // Pack weights into bf16 WMMA B-fragment order
  k_pack_w<<<768, 256, 0, stream>>>(W_proj, WprojPk, 768, 256);
  k_pack_w<<<768, 256, 0, stream>>>(W_ih,   WihPk,   256, 768);
  k_pack_w<<<768, 256, 0, stream>>>(W_hh,   WhhPk,   256, 768);
  k_pack_w<<<256, 256, 0, stream>>>(W_q,    WqPk,    256, 256);

  // Stage 1: projection + scatter into node table
  k_update_nodes<<<N_EXPR / 16, 256, 0, stream>>>(prev_nodes, comb, gidx, WprojPk, b_proj, nodes_ws);

  // Stage 2: GRU (16 paths / block, 16 waves)
  k_gru<<<N_PATHS / 16, 512, 0, stream>>>(pno, peo, pidx, nodes_ws, WihPk, WhhPk, b_ih, b_hh, scat);

  // Stage 3: attention / segment softmax
  k_gemm_q<<<N_NODES / 16, 256, 0, stream>>>(nodes_ws, WqPk, q_ws);
  k_scores<<<(N_PATHS * PATH_LEN) / 8, 256, 0, stream>>>(scat, q_ws, pidx, mask, scores, smax_bits);
  k_soft1<<<(N_PATHS * PATH_LEN + 255) / 256, 256, 0, stream>>>(pidx, mask, smax_bits, scores, denom);
  k_soft2<<<N_PATHS * PATH_LEN, 256, 0, stream>>>(scat, pidx, mask, scores, denom, nodes_out);

  // Stage 4: symbol occurrences scatter-add into expression output
  k_sym_scatter<<<N_OCC, 256, 0, stream>>>(sae, sat, sas, syms, expr_out);
}
